// MorphologicalEmbedding_55087250539192
// MI455X (gfx1250) — matrix operand order repaired and checked by reading, not compiled, not run on hardware
//
#include <hip/hip_runtime.h>
#include <math.h>

typedef __attribute__((ext_vector_type(8)))  float  v8f;
typedef __attribute__((ext_vector_type(8)))  __bf16 v8bf;
typedef __attribute__((ext_vector_type(16))) __bf16 v16bf;

#define DIMX   512
#define NROOTS 3000
#define NPATS  21
#define NAFF   100
#define RDIM   128
#define RANKX  16
#define MF     192            // ROOT_DIM + ROOT_DIM/2
#define MFP2   200            // padded bf16 LDS row (400 B -> 100 dwords, conflict-free)
#define HP2    520            // padded bf16 LDS row (1040 B -> 260 dwords, conflict-free)
#define TPB    32             // tokens per block (two 16-row M tiles)

__device__ __forceinline__ float gelu_exact(float x) {
    return 0.5f * x * (1.0f + erff(x * 0.70710678118654752f));
}

__device__ __forceinline__ void splitf(float x, __bf16& h, __bf16& l) {
    __bf16 hh = (__bf16)x;                 // RNE truncation
    h = hh;
    l = (__bf16)(x - (float)hh);           // residual
}

// Build a v16bf A/B fragment from two contiguous 8-element (16 B) chunks.
__device__ __forceinline__ v16bf frag16(const __bf16* p0, const __bf16* p1) {
    v8bf a = *(const v8bf*)p0;
    v8bf b = *(const v8bf*)p1;
    return __builtin_shufflevector(a, b, 0, 1, 2, 3, 4, 5, 6, 7,
                                         8, 9, 10, 11, 12, 13, 14, 15);
}

#define WMMA_BF16(A, B, C) \
    __builtin_amdgcn_wmma_f32_16x16x32_bf16(false, (A), false, (B), (short)0, (C), false, false)

// ---- one-time weight split: fp32 W -> bf16 hi/lo planes (same [n][k] layout) ----
__global__ void wsplit(const float* __restrict__ W, __bf16* __restrict__ hi,
                       __bf16* __restrict__ lo, int n) {
    int i = blockIdx.x * blockDim.x + threadIdx.x;
    if (i < n) {
        float x = W[i];
        __bf16 h, l;
        splitf(x, h, l);
        hi[i] = h;
        lo[i] = l;
    }
}

// t_shared[e][d] = sum_r shared_U[e][r] * shared_V[r][d]   (128x128, once)
__global__ void tshared_precompute(const float* __restrict__ sU,
                                   const float* __restrict__ sV,
                                   float* __restrict__ tS) {
    int idx = blockIdx.x * blockDim.x + threadIdx.x;   // 0..16383
    int e = idx >> 7;
    int d = idx & 127;
    float acc = 0.f;
#pragma unroll
    for (int r = 0; r < RANKX; ++r)
        acc += sU[e * RANKX + r] * sV[r * RDIM + d];
    tS[idx] = acc;
}

__global__ __launch_bounds__(256, 2)
void morph_fused(const int* __restrict__ token_ids,
                 const int* __restrict__ root_ids,
                 const int* __restrict__ pattern_ids,
                 const int* __restrict__ affix_ids,
                 const float* __restrict__ root_w,
                 const float* __restrict__ tU,      // [21][128][16]
                 const float* __restrict__ tV,      // [21][16][128]
                 const float* __restrict__ affix_w, // [101][64]
                 const __bf16* __restrict__ W1h,    // [512][192] bf16 hi
                 const __bf16* __restrict__ W1l,    // [512][192] bf16 lo
                 const float* __restrict__ b1,
                 const __bf16* __restrict__ W2h,    // [512][512] bf16 hi
                 const __bf16* __restrict__ W2l,    // [512][512] bf16 lo
                 const float* __restrict__ b2,
                 const float* __restrict__ bpe_w,   // [16384][512]
                 const float* __restrict__ gate_bias,
                 const float* __restrict__ tS,      // [128][128]
                 float* __restrict__ out) {
    __shared__ float  s_root[TPB][RDIM];       // 16 KB
    __shared__ float  s_inter[TPB][RANKX];     // 2 KB
    __shared__ __bf16 s_mf_h[TPB][MFP2];       // 12.5 KB
    __shared__ __bf16 s_mf_l[TPB][MFP2];       // 12.5 KB
    __shared__ __bf16 s_h_h[TPB][HP2];         // 32.5 KB
    __shared__ __bf16 s_h_l[TPB][HP2];         // 32.5 KB
    __shared__ int s_sroot[TPB], s_pat[TPB], s_aff[TPB], s_tok[TPB], s_morph[TPB];

    const int tid = threadIdx.x;
    const int t0 = blockIdx.x * TPB;

    // ---- per-token metadata ----
    if (tid < TPB) {
        int t = t0 + tid;
        int ri = root_ids[t];
        int pi = pattern_ids[t];
        s_morph[tid] = (ri >= 0) && (pi >= 0);
        s_sroot[tid] = min(max(ri, 0), NROOTS);
        s_pat[tid]   = min(max(pi, 0), NPATS - 1);
        s_aff[tid]   = min(max(affix_ids[t], 0), NAFF);
        s_tok[tid]   = token_ids[t];
    }
    __syncthreads();

    // ---- gather root vectors ----
    for (int i = tid; i < TPB * RDIM; i += 256) {
        int tk = i >> 7, d = i & 127;
        s_root[tk][d] = root_w[(size_t)s_sroot[tk] * RDIM + d];
    }
    __syncthreads();

    // ---- inter[tk][r] = dot(transform_V[pat][r], root_vec) ----
    for (int i = tid; i < TPB * RANKX; i += 256) {
        int tk = i >> 4, r = i & 15;
        const float* vrow = tV + ((size_t)s_pat[tk] * RANKX + r) * RDIM;
        float acc = 0.f;
#pragma unroll 8
        for (int d = 0; d < RDIM; ++d) acc += vrow[d] * s_root[tk][d];
        s_inter[tk][r] = acc;
    }
    __syncthreads();

    // ---- morph_full = [shared_transform + low-rank delta | affix], split to bf16 planes ----
    for (int i = tid; i < TPB * MF; i += 256) {
        int tk = i / MF, d = i - tk * MF;
        float val;
        if (d < RDIM) {
            const float* trow = tS + (size_t)d * RDIM;
            float acc = 0.f;
#pragma unroll 8
            for (int k = 0; k < RDIM; ++k) acc += trow[k] * s_root[tk][k];
            const float* urow = tU + ((size_t)s_pat[tk] * RDIM + d) * RANKX;
#pragma unroll
            for (int r = 0; r < RANKX; ++r) acc += urow[r] * s_inter[tk][r];
            val = acc;
        } else {
            val = affix_w[(size_t)s_aff[tk] * (RDIM / 2) + (d - RDIM)];
        }
        __bf16 h, l;
        splitf(val, h, l);
        s_mf_h[tk][d] = h;
        s_mf_l[tk][d] = l;
    }
    __syncthreads();

    const int lane = tid & 31;
    const int wv   = tid >> 5;      // wave 0..7
    const int ln   = lane & 15;     // tile column (A: row m, B: col n)
    const int lg   = lane >> 4;     // lane group

    // ---- GEMM1: H = gelu(morph_full @ W1^T + b1), v_wmma_f32_16x16x32_bf16 (split hi/lo) ----
    for (int nt = wv; nt < DIMX / 16; nt += 8) {
        const int nb = nt * 16;
        v8f c0 = {0.f, 0.f, 0.f, 0.f, 0.f, 0.f, 0.f, 0.f};
        v8f c1 = {0.f, 0.f, 0.f, 0.f, 0.f, 0.f, 0.f, 0.f};
        const __bf16* w1hr = W1h + (size_t)(nb + ln) * MF;
        const __bf16* w1lr = W1l + (size_t)(nb + ln) * MF;
#pragma unroll
        for (int kb = 0; kb < MF; kb += 32) {
            // B fragment: chunks K in [kb+8g, +8) and [kb+16+8g, +8) of column n
            v16bf bh = frag16(w1hr + kb + lg * 8, w1hr + kb + 16 + lg * 8);
            v16bf bl = frag16(w1lr + kb + lg * 8, w1lr + kb + 16 + lg * 8);
            // A fragments: chunks K in [kb+8g, +8) and [kb+16+8g, +8)
            v16bf a0h = frag16(&s_mf_h[ln][kb + lg * 8],      &s_mf_h[ln][kb + 16 + lg * 8]);
            v16bf a0l = frag16(&s_mf_l[ln][kb + lg * 8],      &s_mf_l[ln][kb + 16 + lg * 8]);
            v16bf a1h = frag16(&s_mf_h[16 + ln][kb + lg * 8], &s_mf_h[16 + ln][kb + 16 + lg * 8]);
            v16bf a1l = frag16(&s_mf_l[16 + ln][kb + lg * 8], &s_mf_l[16 + ln][kb + 16 + lg * 8]);
            c0 = WMMA_BF16(a0h, bh, c0);
            c0 = WMMA_BF16(a0l, bh, c0);
            c0 = WMMA_BF16(a0h, bl, c0);
            c1 = WMMA_BF16(a1h, bh, c1);
            c1 = WMMA_BF16(a1l, bh, c1);
            c1 = WMMA_BF16(a1h, bl, c1);
        }
        const float bias = b1[nb + ln];
#pragma unroll
        for (int v = 0; v < 8; ++v) {
            const int m = v + 8 * lg;                  // C/D: M = v + 8*group, N = ln
            __bf16 h, l;
            splitf(gelu_exact(c0[v] + bias), h, l);
            s_h_h[m][nb + ln] = h;
            s_h_l[m][nb + ln] = l;
            splitf(gelu_exact(c1[v] + bias), h, l);
            s_h_h[16 + m][nb + ln] = h;
            s_h_l[16 + m][nb + ln] = l;
        }
    }
    __syncthreads();

    const float gate = 1.0f / (1.0f + expf(-gate_bias[0]));
    const float omg  = 1.0f - gate;

    // ---- GEMM2: morph_emb = H @ W2^T + b2, then gated blend with bpe gather ----
    for (int nt = wv; nt < DIMX / 16; nt += 8) {
        const int nb = nt * 16;
        v8f c0 = {0.f, 0.f, 0.f, 0.f, 0.f, 0.f, 0.f, 0.f};
        v8f c1 = {0.f, 0.f, 0.f, 0.f, 0.f, 0.f, 0.f, 0.f};
        const __bf16* w2hr = W2h + (size_t)(nb + ln) * DIMX;
        const __bf16* w2lr = W2l + (size_t)(nb + ln) * DIMX;
#pragma unroll 2
        for (int kb = 0; kb < DIMX; kb += 32) {
            v16bf bh = frag16(w2hr + kb + lg * 8, w2hr + kb + 16 + lg * 8);
            v16bf bl = frag16(w2lr + kb + lg * 8, w2lr + kb + 16 + lg * 8);
            v16bf a0h = frag16(&s_h_h[ln][kb + lg * 8],      &s_h_h[ln][kb + 16 + lg * 8]);
            v16bf a0l = frag16(&s_h_l[ln][kb + lg * 8],      &s_h_l[ln][kb + 16 + lg * 8]);
            v16bf a1h = frag16(&s_h_h[16 + ln][kb + lg * 8], &s_h_h[16 + ln][kb + 16 + lg * 8]);
            v16bf a1l = frag16(&s_h_l[16 + ln][kb + lg * 8], &s_h_l[16 + ln][kb + 16 + lg * 8]);
            c0 = WMMA_BF16(a0h, bh, c0);
            c0 = WMMA_BF16(a0l, bh, c0);
            c0 = WMMA_BF16(a0h, bl, c0);
            c1 = WMMA_BF16(a1h, bh, c1);
            c1 = WMMA_BF16(a1l, bh, c1);
            c1 = WMMA_BF16(a1h, bl, c1);
        }
        const float bias = b2[nb + ln];
#pragma unroll
        for (int v = 0; v < 8; ++v) {
            const int m0 = v + 8 * lg;
            {
                const int tk = m0;
                const float me  = c0[v] + bias;
                const float bpe = bpe_w[(size_t)s_tok[tk] * DIMX + nb + ln];
                const float r   = s_morph[tk] ? (gate * me + omg * bpe) : bpe;
                out[((size_t)(t0 + tk)) * DIMX + nb + ln] = r;
            }
            {
                const int tk = 16 + m0;
                const float me  = c1[v] + bias;
                const float bpe = bpe_w[(size_t)s_tok[tk] * DIMX + nb + ln];
                const float r   = s_morph[tk] ? (gate * me + omg * bpe) : bpe;
                out[((size_t)(t0 + tk)) * DIMX + nb + ln] = r;
            }
        }
    }
}

extern "C" void kernel_launch(void* const* d_in, const int* in_sizes, int n_in,
                              void* d_out, int out_size, void* d_ws, size_t ws_size,
                              hipStream_t stream) {
    const int*   token_ids   = (const int*)d_in[0];
    const int*   root_ids    = (const int*)d_in[1];
    const int*   pattern_ids = (const int*)d_in[2];
    const int*   affix_ids   = (const int*)d_in[3];
    const float* root_w      = (const float*)d_in[4];
    const float* transform_U = (const float*)d_in[5];
    const float* transform_V = (const float*)d_in[6];
    const float* shared_U    = (const float*)d_in[7];
    const float* shared_V    = (const float*)d_in[8];
    const float* affix_w     = (const float*)d_in[9];
    const float* W1          = (const float*)d_in[10];
    const float* b1          = (const float*)d_in[11];
    const float* W2          = (const float*)d_in[12];
    const float* b2          = (const float*)d_in[13];
    const float* bpe_w       = (const float*)d_in[14];
    const float* gate_bias   = (const float*)d_in[15];
    float*       out         = (float*)d_out;

    const int NW1 = DIMX * MF;     // 98304
    const int NW2 = DIMX * DIMX;   // 262144

    // workspace layout (all 16B aligned): tS | W1h | W1l | W2h | W2l  (~1.5 MB)
    char* ws = (char*)d_ws;
    float*  tS  = (float*)ws;                                  // 64 KB
    __bf16* W1h = (__bf16*)(ws + 65536);
    __bf16* W1l = W1h + NW1;
    __bf16* W2h = W1l + NW1;
    __bf16* W2l = W2h + NW2;

    wsplit<<<(NW1 + 255) / 256, 256, 0, stream>>>(W1, W1h, W1l, NW1);
    wsplit<<<(NW2 + 255) / 256, 256, 0, stream>>>(W2, W2h, W2l, NW2);
    tshared_precompute<<<64, 256, 0, stream>>>(shared_U, shared_V, tS);

    const int T = in_sizes[0];  // B*S = 32768 (divisible by TPB)
    morph_fused<<<T / TPB, 256, 0, stream>>>(
        token_ids, root_ids, pattern_ids, affix_ids,
        root_w, transform_U, transform_V, affix_w,
        W1h, W1l, b1, W2h, W2l, b2, bpe_w, gate_bias, tS, out);
}
